// MultiHeadSelfAttention_40252433498196
// MI455X (gfx1250) — matrix operand order repaired and checked by reading, not compiled, not run on hardware
//
#include <hip/hip_runtime.h>

// B=2, N=2048, E=1024, H=16, D=64. Output f32 [B,N,E].
// Workspace (bf16): Xq,Xk,Xv [4096,1024] + Wq,Wk,Wv,Wo [1024,1024]
//                 + Qh,Kh,Vh,Oh [32][2048][64]  => 32 Mi elements = 64 MiB.

typedef __attribute__((ext_vector_type(16))) __bf16 v16bf;
typedef __attribute__((ext_vector_type(8)))  __bf16 v8bf;
typedef __attribute__((ext_vector_type(8)))  float  v8f;

#define NSEQ 2048
#define EMB  1024
#define NBH  32    // B*H
#define HDIM 64
#define MROWS 4096 // B*N

__device__ __forceinline__ v8f wmma_bf16(v16bf a, v16bf b, v8f c) {
  return __builtin_amdgcn_wmma_f32_16x16x32_bf16(false, a, false, b, (short)0, c, false, false);
}

// A fragment (16x32 bf16): per-lane (row = lane&15): K = 16*half + e, contiguous.
__device__ __forceinline__ v16bf load_a_bf16(const __bf16* __restrict__ row, int k0, int half) {
  v8bf lo = *(const v8bf*)(row + k0 + 16 * half);
  v8bf hi = *(const v8bf*)(row + k0 + 16 * half + 8);
  v16bf a;
#pragma unroll
  for (int j = 0; j < 8; ++j) { a[j] = lo[j]; a[8 + j] = hi[j]; }
  return a;
}

// B fragment (32x16 bf16) with B = W^T, W row-major [n][k]: per-lane K = 16*half + e, contiguous.
__device__ __forceinline__ v16bf load_b_bf16(const __bf16* __restrict__ wrow, int k0, int half) {
  return *(const v16bf*)(wrow + k0 + 16 * half);
}

__device__ __forceinline__ float rowmax16(float v) {
#pragma unroll
  for (int m = 1; m < 16; m <<= 1) v = fmaxf(v, __shfl_xor(v, m, 16));
  return v;
}
__device__ __forceinline__ float rowsum16(float v) {
#pragma unroll
  for (int m = 1; m < 16; m <<= 1) v += __shfl_xor(v, m, 16);
  return v;
}

// ---------------------------------------------------------------------------
// Kernel 0: f32 -> bf16 conversion, 8 elements/thread.
// ---------------------------------------------------------------------------
__global__ __launch_bounds__(256) void cvt_f32_bf16(const float* __restrict__ in,
                                                    __bf16* __restrict__ out, int n8) {
  int i = blockIdx.x * 256 + threadIdx.x;
  if (i < n8) {
    float4 f0 = ((const float4*)in)[2 * i];
    float4 f1 = ((const float4*)in)[2 * i + 1];
    v8bf o;
    o[0] = (__bf16)f0.x; o[1] = (__bf16)f0.y; o[2] = (__bf16)f0.z; o[3] = (__bf16)f0.w;
    o[4] = (__bf16)f1.x; o[5] = (__bf16)f1.y; o[6] = (__bf16)f1.z; o[7] = (__bf16)f1.w;
    ((v8bf*)out)[i] = o;
  }
}

// ---------------------------------------------------------------------------
// Kernel 1: Y[bh][n][d] (bf16) = X[4096,1024]bf16 @ W^T(bf16), head-split out.
// Block 256 thr (8 waves as 4x2), block tile 128x128, wave tile 32x64.
// ---------------------------------------------------------------------------
__global__ __launch_bounds__(256) void gemm_proj(const __bf16* __restrict__ X,
                                                 const __bf16* __restrict__ W,
                                                 __bf16* __restrict__ Y) {
  const int lane = threadIdx.x & 31;
  const int w    = threadIdx.x >> 5;
  const int half = lane >> 4, ln = lane & 15;
  const int mw = w >> 1, nw = w & 1;
  const int m0 = blockIdx.y * 128 + mw * 32;
  const int n0 = blockIdx.x * 128 + nw * 64;

  const __bf16* xr[2];
  const __bf16* wr[4];
#pragma unroll
  for (int i = 0; i < 2; ++i) xr[i] = X + (size_t)(m0 + 16 * i + ln) * EMB;
#pragma unroll
  for (int j = 0; j < 4; ++j) wr[j] = W + (size_t)(n0 + 16 * j + ln) * EMB;

  v8f acc[2][4] = {};
  for (int k0 = 0; k0 < EMB; k0 += 32) {
    if (k0 + 64 < EMB) {
      __builtin_prefetch(xr[0] + k0 + 64, 0, 1);
      __builtin_prefetch(wr[0] + k0 + 64, 0, 1);
    }
    v16bf a[2], b[4];
#pragma unroll
    for (int i = 0; i < 2; ++i) a[i] = load_a_bf16(xr[i], k0, half);
#pragma unroll
    for (int j = 0; j < 4; ++j) b[j] = load_b_bf16(wr[j], k0, half);
#pragma unroll
    for (int i = 0; i < 2; ++i)
#pragma unroll
      for (int j = 0; j < 4; ++j) acc[i][j] = wmma_bf16(a[i], b[j], acc[i][j]);
  }

  // D layout: (r, lane) -> row = r + 8*half, col = lane&15. Scatter to [bh][n][d].
#pragma unroll
  for (int i = 0; i < 2; ++i)
#pragma unroll
    for (int j = 0; j < 4; ++j)
#pragma unroll
      for (int r = 0; r < 8; ++r) {
        int m = m0 + 16 * i + r + 8 * half;      // global row (b*N + n)
        int c = n0 + 16 * j + ln;                // global col (h*D + d)
        int b_ = m >> 11, nn = m & 2047;
        int h = c >> 6,   d  = c & 63;
        Y[((size_t)(b_ * 16 + h) * NSEQ + nn) * HDIM + d] = (__bf16)acc[i][j][r];
      }
}

// ---------------------------------------------------------------------------
// Kernel 2: flash attention per (b,h). Grid (N/128, B*H), block 256.
// K tile staged with global_load_async_to_lds_b128 (ASYNCcnt), V transposed manually.
// ---------------------------------------------------------------------------
__global__ __launch_bounds__(256) void flash_attn(const __bf16* __restrict__ Qh,
                                                  const __bf16* __restrict__ Kh,
                                                  const __bf16* __restrict__ Vh,
                                                  __bf16* __restrict__ Oh) {
  __shared__ __align__(32) __bf16 sK[32 * 64];        // K tile, row-major [key][d]
  __shared__ __align__(32) __bf16 sVt[64 * 32];       // V tile transposed [d][key]
  __shared__ __align__(32) __bf16 sP[8][16 * 32];     // per-wave P tile [row][key]

  const int tid  = threadIdx.x;
  const int lane = tid & 31, w = tid >> 5;
  const int half = lane >> 4, ln = lane & 15;
  const int bh = blockIdx.y;
  const int q0 = blockIdx.x * 128 + w * 16;

  const __bf16* Qb = Qh + (size_t)bh * NSEQ * HDIM;
  const __bf16* Kb = Kh + (size_t)bh * NSEQ * HDIM;
  const __bf16* Vb = Vh + (size_t)bh * NSEQ * HDIM;

  // Q fragments: A-frags for d 0..31 and 32..63, rows q0..q0+15.
  v16bf aq[2];
  {
    const __bf16* qrow = Qb + (size_t)(q0 + ln) * HDIM;
#pragma unroll
    for (int t = 0; t < 2; ++t) {
      v8bf lo = *(const v8bf*)(qrow + 32 * t + 16 * half);
      v8bf hi = *(const v8bf*)(qrow + 32 * t + 16 * half + 8);
#pragma unroll
      for (int j = 0; j < 8; ++j) { aq[t][j] = lo[j]; aq[t][8 + j] = hi[j]; }
    }
  }

  float mrow[8], lrow[8];
  v8f o[4] = {};
#pragma unroll
  for (int r = 0; r < 8; ++r) { mrow[r] = -3.0e38f; lrow[r] = 0.0f; }

  const int srow = tid >> 3;          // 0..31 staging row
  const int scol = (tid & 7) * 8;     // 0..56 staging col base (16B chunks)
  const unsigned ldsK = (unsigned)(size_t)&sK[srow * 64 + scol];   // LDS addr = low 32 bits

  for (int kt = 0; kt < NSEQ; kt += 32) {
    // ---- async-copy K tile to LDS (one b128 per lane, 4 KiB total) ----
    {
      unsigned long long gK =
          (unsigned long long)(size_t)(Kb + (size_t)(kt + srow) * HDIM + scol);
      asm volatile("global_load_async_to_lds_b128 %0, %1, off"
                   :: "v"(ldsK), "v"(gK) : "memory");
    }
    // ---- V tile through registers, transposed into LDS ----
    v8bf vv = *(const v8bf*)(Vb + (size_t)(kt + srow) * HDIM + scol);
#pragma unroll
    for (int j = 0; j < 8; ++j) sVt[(scol + j) * 32 + srow] = vv[j];
    asm volatile("s_wait_asynccnt 0x0" ::: "memory");
    __syncthreads();

    // ---- S = Q K^T for two 16-key subtiles ----
    v8f s[2] = {};
#pragma unroll
    for (int n = 0; n < 2; ++n) {
      v16bf bk0 = *(const v16bf*)&sK[(n * 16 + ln) * 64 + 16 * half];        // d 0..31
      v16bf bk1 = *(const v16bf*)&sK[(n * 16 + ln) * 64 + 32 + 16 * half];   // d 32..63
      s[n] = wmma_bf16(aq[0], bk0, s[n]);
      s[n] = wmma_bf16(aq[1], bk1, s[n]);
    }
#pragma unroll
    for (int r = 0; r < 8; ++r) { s[0][r] *= 0.03125f; s[1][r] *= 0.03125f; } // /sqrt(E)

    // ---- online softmax (row = r + 8*half; reductions within 16-lane group) ----
    float alpha[8];
#pragma unroll
    for (int r = 0; r < 8; ++r) {
      float t = rowmax16(fmaxf(s[0][r], s[1][r]));
      float mnew = fmaxf(mrow[r], t);
      float a_ = __expf(mrow[r] - mnew);
      mrow[r] = mnew;
      float p0 = __expf(s[0][r] - mnew);
      float p1 = __expf(s[1][r] - mnew);
      s[0][r] = p0; s[1][r] = p1;
      lrow[r] = lrow[r] * a_ + rowsum16(p0 + p1);
      alpha[r] = a_;
    }
#pragma unroll
    for (int dt = 0; dt < 4; ++dt)
#pragma unroll
      for (int r = 0; r < 8; ++r) o[dt][r] *= alpha[r];

    // ---- D-layout -> A-layout via per-wave LDS round trip ----
    __bf16* myP = &sP[w][0];
#pragma unroll
    for (int r = 0; r < 8; ++r) {
      int row = r + 8 * half;
      myP[row * 32 + ln]      = (__bf16)s[0][r];
      myP[row * 32 + 16 + ln] = (__bf16)s[1][r];
    }
    v16bf pa;
    {
      v8bf lo = *(const v8bf*)&myP[ln * 32 + 16 * half];
      v8bf hi = *(const v8bf*)&myP[ln * 32 + 16 * half + 8];
#pragma unroll
      for (int j = 0; j < 8; ++j) { pa[j] = lo[j]; pa[8 + j] = hi[j]; }
    }

    // ---- O += P V (B-frags contiguous from transposed V) ----
#pragma unroll
    for (int dt = 0; dt < 4; ++dt) {
      v16bf bv = *(const v16bf*)&sVt[(dt * 16 + ln) * 32 + 16 * half];
      o[dt] = wmma_bf16(pa, bv, o[dt]);
    }
    __syncthreads();
  }

  // ---- normalize + store ----
  float invl[8];
#pragma unroll
  for (int r = 0; r < 8; ++r) invl[r] = 1.0f / lrow[r];
  __bf16* Ob = Oh + (size_t)bh * NSEQ * HDIM;
#pragma unroll
  for (int dt = 0; dt < 4; ++dt)
#pragma unroll
    for (int r = 0; r < 8; ++r) {
      int row = q0 + r + 8 * half;
      int d = dt * 16 + ln;
      Ob[(size_t)row * HDIM + d] = (__bf16)(o[dt][r] * invl[r]);
    }
}

// ---------------------------------------------------------------------------
// Kernel 3: out[4096,1024] f32 = attn([B,H,N,D] bf16) @ Wo^T (bf16).
// Same tiling as gemm_proj; A-fragments gathered through the (h,d) mapping.
// ---------------------------------------------------------------------------
__global__ __launch_bounds__(256) void gemm_out(const __bf16* __restrict__ Oh,
                                                const __bf16* __restrict__ Wo,
                                                float* __restrict__ out) {
  const int lane = threadIdx.x & 31;
  const int w    = threadIdx.x >> 5;
  const int half = lane >> 4, ln = lane & 15;
  const int mw = w >> 1, nw = w & 1;
  const int m0 = blockIdx.y * 128 + mw * 32;
  const int n0 = blockIdx.x * 128 + nw * 64;

  const __bf16* wr[4];
#pragma unroll
  for (int j = 0; j < 4; ++j) wr[j] = Wo + (size_t)(n0 + 16 * j + ln) * EMB;

  v8f acc[2][4] = {};
  for (int k0 = 0; k0 < EMB; k0 += 32) {
    v16bf a[2], b[4];
#pragma unroll
    for (int i = 0; i < 2; ++i) {
      int m = m0 + 16 * i + ln;
      int b_ = m >> 11, nn = m & 2047;
      int ka = k0 + 16 * half;
      int kb = k0 + 16 * half + 8;
      const __bf16* pa = Oh + ((size_t)(b_ * 16 + (ka >> 6)) * NSEQ + nn) * HDIM + (ka & 63);
      const __bf16* pb = Oh + ((size_t)(b_ * 16 + (kb >> 6)) * NSEQ + nn) * HDIM + (kb & 63);
      v8bf lo = *(const v8bf*)pa;
      v8bf hi = *(const v8bf*)pb;
#pragma unroll
      for (int j = 0; j < 8; ++j) { a[i][j] = lo[j]; a[i][8 + j] = hi[j]; }
    }
#pragma unroll
    for (int j = 0; j < 4; ++j) b[j] = load_b_bf16(wr[j], k0, half);
#pragma unroll
    for (int i = 0; i < 2; ++i)
#pragma unroll
      for (int j = 0; j < 4; ++j) acc[i][j] = wmma_bf16(a[i], b[j], acc[i][j]);
  }

#pragma unroll
  for (int i = 0; i < 2; ++i)
#pragma unroll
    for (int j = 0; j < 4; ++j)
#pragma unroll
      for (int r = 0; r < 8; ++r) {
        int m = m0 + 16 * i + r + 8 * half;
        int c = n0 + 16 * j + ln;
        out[(size_t)m * EMB + c] = acc[i][j][r];
      }
}

extern "C" void kernel_launch(void* const* d_in, const int* in_sizes, int n_in,
                              void* d_out, int out_size, void* d_ws, size_t ws_size,
                              hipStream_t stream) {
  const float* q  = (const float*)d_in[0];
  const float* k  = (const float*)d_in[1];
  const float* v  = (const float*)d_in[2];
  const float* Wq = (const float*)d_in[3];
  const float* Wk = (const float*)d_in[4];
  const float* Wv = (const float*)d_in[5];
  const float* Wo = (const float*)d_in[6];
  float* out = (float*)d_out;

  const size_t nX = (size_t)MROWS * EMB;        // 4 Mi
  const size_t nW = (size_t)EMB * EMB;          // 1 Mi
  const size_t nH = (size_t)NBH * NSEQ * HDIM;  // 4 Mi

  __bf16* p   = (__bf16*)d_ws;
  __bf16* Xq  = p; p += nX;
  __bf16* Xk  = p; p += nX;
  __bf16* Xv  = p; p += nX;
  __bf16* Wqb = p; p += nW;
  __bf16* Wkb = p; p += nW;
  __bf16* Wvb = p; p += nW;
  __bf16* Wob = p; p += nW;
  __bf16* Qh  = p; p += nH;
  __bf16* Kh  = p; p += nH;
  __bf16* Vh  = p; p += nH;
  __bf16* Oh  = p;                              // total 64 MiB

  dim3 blk(256);
  cvt_f32_bf16<<<dim3(nX / 8 / 256), blk, 0, stream>>>(q,  Xq,  (int)(nX / 8));
  cvt_f32_bf16<<<dim3(nX / 8 / 256), blk, 0, stream>>>(k,  Xk,  (int)(nX / 8));
  cvt_f32_bf16<<<dim3(nX / 8 / 256), blk, 0, stream>>>(v,  Xv,  (int)(nX / 8));
  cvt_f32_bf16<<<dim3(nW / 8 / 256), blk, 0, stream>>>(Wq, Wqb, (int)(nW / 8));
  cvt_f32_bf16<<<dim3(nW / 8 / 256), blk, 0, stream>>>(Wk, Wkb, (int)(nW / 8));
  cvt_f32_bf16<<<dim3(nW / 8 / 256), blk, 0, stream>>>(Wv, Wvb, (int)(nW / 8));
  cvt_f32_bf16<<<dim3(nW / 8 / 256), blk, 0, stream>>>(Wo, Wob, (int)(nW / 8));

  dim3 gGemm(EMB / 128, MROWS / 128);           // 8 x 32
  gemm_proj<<<gGemm, blk, 0, stream>>>(Xq, Wqb, Qh);
  gemm_proj<<<gGemm, blk, 0, stream>>>(Xk, Wkb, Kh);
  gemm_proj<<<gGemm, blk, 0, stream>>>(Xv, Wvb, Vh);

  dim3 gAttn(NSEQ / 128, NBH);                  // 16 x 32
  flash_attn<<<gAttn, blk, 0, stream>>>(Qh, Kh, Vh, Oh);

  gemm_out<<<gGemm, blk, 0, stream>>>(Oh, Wob, out);
}
